// GenericGraphEncoder_51084341019403
// MI455X (gfx1250) — matrix operand order repaired and checked by reading, not compiled, not run on hardware
//
#include <hip/hip_runtime.h>
#include <hip/hip_bf16.h>
#include <stdint.h>

// ---------------- problem constants (from reference) ----------------
static constexpr int kN = 50000;     // nodes
static constexpr int kE = 800000;    // edges
static constexpr int kB = 512;       // graphs
static constexpr int kR = 3;         // relations
static constexpr int kDIN = 128;
static constexpr int kH = 64;
static constexpr int kL = 12;
static constexpr int kNodeDim = kH * (kL + 1);   // 832
static constexpr int kGRS = kNodeDim / 2;        // 416
static constexpr int kHeads = 8;
static constexpr int kChunk = kGRS / kHeads;     // 52
static constexpr int kYW = 768;                  // FiLM fused GEMM width: 64+128+3*64+3*128

typedef __bf16 bf16_t;
typedef __attribute__((ext_vector_type(16))) __bf16 v16bf;
typedef __attribute__((ext_vector_type(8)))  __bf16 v8bf;
typedef __attribute__((ext_vector_type(8)))  float  v8f;

// ---------------- small device helpers ----------------
__device__ __forceinline__ float leaky(float x) { return x > 0.f ? x : 0.01f * x; }

__device__ __forceinline__ bf16_t f2bf(float f) {
  unsigned u = __float_as_uint(f);
  unsigned r = u + 0x7FFFu + ((u >> 16) & 1u);   // round-to-nearest-even
  unsigned short h = (unsigned short)(r >> 16);
  return __builtin_bit_cast(bf16_t, h);
}

__device__ __forceinline__ void atomicMaxF(float* a, float v) {
  unsigned* ua = (unsigned*)a;
  unsigned old = *ua;
  while (__uint_as_float(old) < v) {
    unsigned prev = atomicCAS(ua, old, __float_as_uint(v));
    if (prev == old) break;
    old = prev;
  }
}

// ---------------- generic kernels ----------------
__global__ void fill_k(float* __restrict__ p, float v, int n) {
  int i = blockIdx.x * blockDim.x + threadIdx.x;
  if (i < n) p[i] = v;
}

__global__ void cast_f32_bf16_k(const float* __restrict__ s, bf16_t* __restrict__ d, int n) {
  int i = blockIdx.x * blockDim.x + threadIdx.x;
  if (i < n) d[i] = f2bf(s[i]);
}

// Cast weight matrix src (rows x srccols, row-major = K x N) into TRANSPOSED bf16 storage
// dst[(colofs + c) * ldt + r]  (N-major), zero-padding columns >= srccols.
// N-major B makes each WMMA B-fragment two contiguous 128-bit loads per lane.
__global__ void copy_cast_wT_k(const float* __restrict__ src, int rows, int srccols,
                               bf16_t* __restrict__ dst, int ldt, int colofs, int dstcols) {
  int i = blockIdx.x * blockDim.x + threadIdx.x;
  int total = rows * dstcols;
  if (i >= total) return;
  int r = i / dstcols, c = i - r * dstcols;
  float v = (c < srccols) ? src[(size_t)r * srccols + c] : 0.0f;
  dst[(size_t)(colofs + c) * ldt + r] = f2bf(v);
}

// ---------------- WMMA bf16 GEMM: C(f32) = A(bf16, row-major) * B(bf16, N-major/transposed) ----
// One wave computes a 16(M) x 32(N) tile: one A fragment feeds two accumulators.
// K stepped by 32 (unroll 2). M % 16 == 0, Ncol % 32 == 0, (K/32) % 2 == 0.
// Fragments use contiguous k order (k = 16*hi + s) in A and B (wave32). All fragment loads are 128-bit.
__global__ void gemm_bf16_wmma_k(const bf16_t* __restrict__ A, int lda,
                                 const bf16_t* __restrict__ Bt, int ldt,
                                 float* __restrict__ C, int ldc,
                                 int Ncol, int K, int tilesN, int totalTiles) {
  int wave = blockIdx.x * (blockDim.x >> 5) + (threadIdx.x >> 5);
  if (wave >= totalTiles) return;
  int lane = threadIdx.x & 31;
  int tm = wave / tilesN;
  int tn = wave - tm * tilesN;
  int hi  = lane >> 4;       // lane group 0/1
  int l15 = lane & 15;

  // A: lane holds row M = l15; K sub-block 0..15 (hi=0) or 16..31 (hi=1)
  const bf16_t* arow  = A  + (size_t)(tm * 16 + l15) * lda + hi * 16;
  // B: lane holds col N = l15 (+16 for frag1); K sub-block 0..15 (hi=0) or 16..31 (hi=1)
  const bf16_t* brow0 = Bt + (size_t)(tn * 32 + l15) * ldt + hi * 16;
  const bf16_t* brow1 = brow0 + (size_t)16 * ldt;

  v8f acc0 = {}, acc1 = {};
#pragma unroll 2
  for (int k = 0; k < K; k += 32) {
    // near-cache (WGP-scope) prefetch of the fragments two K-steps ahead
    __builtin_prefetch(arow + k + 64, 0, 3);
    __builtin_prefetch(brow0 + k + 64, 0, 3);
    __builtin_prefetch(brow1 + k + 64, 0, 3);
    v8bf a0  = *(const v8bf*)(arow + k);
    v8bf a1  = *(const v8bf*)(arow + k + 8);
    v8bf b00 = *(const v8bf*)(brow0 + k);
    v8bf b01 = *(const v8bf*)(brow0 + k + 8);
    v8bf b10 = *(const v8bf*)(brow1 + k);
    v8bf b11 = *(const v8bf*)(brow1 + k + 8);
    v16bf av  = __builtin_shufflevector(a0, a1, 0,1,2,3,4,5,6,7,8,9,10,11,12,13,14,15);
    v16bf bv0 = __builtin_shufflevector(b00, b01, 0,1,2,3,4,5,6,7,8,9,10,11,12,13,14,15);
    v16bf bv1 = __builtin_shufflevector(b10, b11, 0,1,2,3,4,5,6,7,8,9,10,11,12,13,14,15);
    acc0 = __builtin_amdgcn_wmma_f32_16x16x32_bf16(false, av, false, bv0, (short)0, acc0,
                                                   false, false);
    acc1 = __builtin_amdgcn_wmma_f32_16x16x32_bf16(false, av, false, bv1, (short)0, acc1,
                                                   false, false);
  }
  // C/D layout: VGPR v -> row tm*16 + hi*8 + v, col = l15 within each 16-col group
  float* cp = C + (size_t)(tm * 16 + hi * 8) * ldc + tn * 32 + l15;
#pragma unroll
  for (int v = 0; v < 8; ++v) {
    cp[(size_t)v * ldc]      = acc0[v];
    cp[(size_t)v * ldc + 16] = acc1[v];
  }
}

// ---------------- FiLM kernels ----------------
// Y layout per node (768 cols): [0,64)=x@W_skip  [64,128)=beta_s  [128,192)=gamma_s
//                               [192+64r, ..)=h_r  [384+128r, +64)=beta_r  (+64..128)=gamma_r
__global__ void film_post_k(const float* __restrict__ Y, float* __restrict__ nr, int colofs) {
  int i = blockIdx.x * blockDim.x + threadIdx.x;
  if (i >= kN * kH) return;
  int n = i >> 6, d = i & 63;
  const float* y = Y + (size_t)n * kYW;
  nr[(size_t)n * kNodeDim + colofs + d] = leaky(y[128 + d] * y[d] + y[64 + d]);
}

__global__ void edge_msg_k(const int* __restrict__ ei, const int* __restrict__ et,
                           const float* __restrict__ Y, const float* __restrict__ Fb,
                           float* __restrict__ sums, float* __restrict__ cnt) {
  int t = threadIdx.x;
  int e = blockIdx.x * 4 + (t >> 6);
  if (e >= kE) return;
  int d  = t & 63;
  int s  = ei[e];
  int dn = ei[kE + e];
  int r  = et[e];
  float hv = Y[(size_t)s * kYW + 192 + r * 64 + d];
  float be = Y[(size_t)dn * kYW + 384 + r * 128 + d] + Fb[r * 128 + d];
  float ga = Y[(size_t)dn * kYW + 384 + r * 128 + 64 + d] + Fb[r * 128 + 64 + d];
  float m = leaky(ga * hv + be);
  atomicAdd(&sums[((size_t)dn * kR + r) * 64 + d], m);
  if (d == 0) atomicAdd(&cnt[dn * kR + r], 1.0f);
}

__global__ void film_mean_k(const float* __restrict__ sums, const float* __restrict__ cnt,
                            float* __restrict__ nr, int colofs) {
  int i = blockIdx.x * blockDim.x + threadIdx.x;
  if (i >= kN * kH) return;
  int n = i >> 6, d = i & 63;
  float acc = nr[(size_t)n * kNodeDim + colofs + d];
#pragma unroll
  for (int r = 0; r < kR; ++r) {
    float c = cnt[n * kR + r];
    acc += sums[((size_t)n * kR + r) * 64 + d] / fmaxf(c, 1.0f);
  }
  nr[(size_t)n * kNodeDim + colofs + d] = acc;
}

// global (whole-matrix) LN: reduce then apply
__global__ void reduce_sum_sq_k(const float* __restrict__ nr, int colofs, float* __restrict__ red) {
  __shared__ float s0[256], s1[256];
  float a = 0.f, b = 0.f;
  int total = kN * kH;
  for (int i = blockIdx.x * blockDim.x + threadIdx.x; i < total; i += gridDim.x * blockDim.x) {
    int n = i >> 6, d = i & 63;
    float v = nr[(size_t)n * kNodeDim + colofs + d];
    a += v; b += v * v;
  }
  s0[threadIdx.x] = a; s1[threadIdx.x] = b;
  __syncthreads();
  for (int st = 128; st > 0; st >>= 1) {
    if (threadIdx.x < st) { s0[threadIdx.x] += s0[threadIdx.x + st]; s1[threadIdx.x] += s1[threadIdx.x + st]; }
    __syncthreads();
  }
  if (threadIdx.x == 0) { atomicAdd(&red[0], s0[0]); atomicAdd(&red[1], s1[0]); }
}

__global__ void ln_apply_k(const float* __restrict__ nr, int colofs,
                           const float* __restrict__ w, const float* __restrict__ b,
                           const float* __restrict__ red, bf16_t* __restrict__ xb) {
  int i = blockIdx.x * blockDim.x + threadIdx.x;
  if (i >= kN * kH) return;
  int n = i >> 6, d = i & 63;
  const float inv = 1.0f / (float)(kN * kH);
  float mean = red[0] * inv;
  float var  = fmaxf(red[1] * inv - mean * mean, 0.0f);
  float stdv = sqrtf(var);
  float v = (nr[(size_t)n * kNodeDim + colofs + d] - mean) / (stdv + 1e-5f) * w[d] + b[d];
  xb[(size_t)n * kH + d] = f2bf(v);
}

// ---------------- aggregator kernels ----------------
// Y1 is N x 512 : blocks [s_softmax | t_softmax | s_sigmoid | t_sigmoid]
__global__ void bias_act_cast4_k(const float* __restrict__ Y,
                                 const float* __restrict__ b0, const float* __restrict__ b1,
                                 const float* __restrict__ b2, const float* __restrict__ b3,
                                 bf16_t* __restrict__ dst) {
  int i = blockIdx.x * blockDim.x + threadIdx.x;
  if (i >= kN * 512) return;
  int c = i & 511;
  const float* bs = (c < 128) ? b0 : (c < 256) ? b1 : (c < 384) ? b2 : b3;
  dst[i] = f2bf(leaky(Y[i] + bs[c & 127]));
}

// S is N x 64 : softmax scores at cols [0,8), sigmoid scores at cols [32,40)
__global__ void seg_max_k(const float* __restrict__ S, const float* __restrict__ bias,
                          const int* __restrict__ batch, float* __restrict__ segmax) {
  int i = blockIdx.x * blockDim.x + threadIdx.x;
  if (i >= kN * kHeads) return;
  int n = i >> 3, h = i & 7;
  atomicMaxF(&segmax[batch[n] * kHeads + h], S[(size_t)n * 64 + h] + bias[h]);
}

__global__ void seg_expsum_k(const float* __restrict__ S, const float* __restrict__ bias,
                             const int* __restrict__ batch, const float* __restrict__ segmax,
                             float* __restrict__ wts, float* __restrict__ segsum) {
  int i = blockIdx.x * blockDim.x + threadIdx.x;
  if (i >= kN * kHeads) return;
  int n = i >> 3, h = i & 7;
  float v = S[(size_t)n * 64 + h] + bias[h];
  float w = __expf(v - segmax[batch[n] * kHeads + h]);
  wts[(size_t)n * 16 + h] = w;
  atomicAdd(&segsum[batch[n] * kHeads + h], w);
}

__global__ void seg_norm_k(float* __restrict__ wts, const int* __restrict__ batch,
                           const float* __restrict__ segsum) {
  int i = blockIdx.x * blockDim.x + threadIdx.x;
  if (i >= kN * kHeads) return;
  int n = i >> 3, h = i & 7;
  wts[(size_t)n * 16 + h] /= segsum[batch[n] * kHeads + h];
}

__global__ void sigmoid_w_k(const float* __restrict__ S, const float* __restrict__ bias,
                            float* __restrict__ wts) {
  int i = blockIdx.x * blockDim.x + threadIdx.x;
  if (i >= kN * kHeads) return;
  int n = i >> 3, h = i & 7;
  float v = S[(size_t)n * 64 + 32 + h] + bias[h];
  wts[(size_t)n * 16 + 8 + h] = 1.0f / (1.0f + __expf(-v));
}

__global__ void scatter_k(const float* __restrict__ T, const float* __restrict__ bias,
                          const float* __restrict__ wts, int wofs,
                          const int* __restrict__ batch, float* __restrict__ out, int colofs) {
  int i = blockIdx.x * blockDim.x + threadIdx.x;
  if (i >= kN * kGRS) return;
  int n = i / kGRS, c = i - n * kGRS;
  int head = c / kChunk;
  float w = wts[(size_t)n * 16 + wofs + head];
  float v = leaky(T[i] + bias[c]);
  atomicAdd(&out[(size_t)batch[n] * (2 * kGRS) + colofs + c], w * v);
}

// ---------------- host orchestration ----------------
static inline int ceil_div(int a, int b) { return (a + b - 1) / b; }

static void gemm(hipStream_t stream, const bf16_t* A, int lda, const bf16_t* Bt, int ldt,
                 float* C, int ldc, int M, int Ncol, int K) {
  int tilesN = Ncol / 32;
  int tiles = (M / 16) * tilesN;
  int blocks = ceil_div(tiles, 4);                 // 4 waves / block (wave32)
  gemm_bf16_wmma_k<<<blocks, 128, 0, stream>>>(A, lda, Bt, ldt, C, ldc, Ncol, K, tilesN, tiles);
}

extern "C" void kernel_launch(void* const* d_in, const int* in_sizes, int n_in,
                              void* d_out, int out_size, void* d_ws, size_t ws_size,
                              hipStream_t stream) {
  (void)in_sizes; (void)n_in; (void)out_size; (void)ws_size;

  const float* x          = (const float*)d_in[0];
  const int*   edge_index = (const int*)d_in[1];
  const int*   edge_type  = (const int*)d_in[2];
  const int*   batch      = (const int*)d_in[3];
  auto fin = [&](int i) { return (const float*)d_in[i]; };
  // params: layers[13]{W_skip,F_skip,W,F,Fb} -> 4..68 ; lns[12]{w,b} -> 69..92
  // softmax_aggr{scoring{Ws0..2,bs0..2},transform{Ws0..2,bs0..2}} -> 93..104 ; sigmoid -> 105..116
  auto layerP = [&](int l, int j) { return fin(4 + 5 * l + j); };
  auto lnP    = [&](int i, int j) { return fin(69 + 2 * i + j); };
  auto aggP   = [&](int a, int t, int j) { return fin(93 + a * 12 + t * 6 + j); };

  float* out = (float*)d_out;

  // ------- workspace carving (phase1 and phase2 scratch overlap) -------
  uintptr_t base = (uintptr_t)d_ws;
  auto al = [](uintptr_t v) { return (v + 255) & ~(uintptr_t)255; };
  uintptr_t p = al(base);
  float* nr  = (float*)p; p = al(p + (size_t)kN * kNodeDim * 4);    // node_repr (persistent)
  float* red = (float*)p; p = al(p + 256);                          // LN reduction scratch
  uintptr_t scbase = p;
  // phase 1 (FiLM layers)
  p = scbase;
  bf16_t* xb   = (bf16_t*)p; p = al(p + (size_t)kN * kDIN * 2);
  bf16_t* Wcat = (bf16_t*)p; p = al(p + (size_t)kYW * kDIN * 2);    // transposed: 768 x din
  float*  Y    = (float*)p;  p = al(p + (size_t)kN * kYW * 4);
  float*  sums = (float*)p;  p = al(p + (size_t)kN * kR * 64 * 4);
  float*  cnt  = (float*)p;  p = al(p + (size_t)kN * kR * 4);
  // phase 2 (pooling) — overlaps phase 1 scratch
  p = scbase;
  bf16_t* nrb  = (bf16_t*)p; float* T = (float*)p; p = al(p + (size_t)kN * kNodeDim * 2);
  bf16_t* Wagg = (bf16_t*)p; p = al(p + (size_t)512 * kNodeDim * 2);  // transposed: 512 x 832 max
  bf16_t* ag1b = (bf16_t*)p; p = al(p + (size_t)kN * 512 * 2);
  bf16_t* ag2b = (bf16_t*)p; p = al(p + (size_t)kN * 512 * 2);
  float*  S    = (float*)p;  p = al(p + (size_t)kN * 64 * 4);
  float*  wts  = (float*)p;  p = al(p + (size_t)kN * 16 * 4);
  float*  segmax = (float*)p; p = al(p + (size_t)kB * kHeads * 4);
  float*  segsum = (float*)p; p = al(p + (size_t)kB * kHeads * 4);
  float*  Y1 = nr;  // node_repr region reused as N x 512 GEMM output after nrb is made

  const int TB = 256;
  auto g1 = [&](int total) { return dim3(ceil_div(total, TB)); };

  // ================= FiLM layers =================
  cast_f32_bf16_k<<<g1(kN * kDIN), TB, 0, stream>>>(x, xb, kN * kDIN);

  for (int l = 0; l <= kL; ++l) {
    int din = (l == 0) ? kDIN : kH;
    const float* Wsk = layerP(l, 0);
    const float* Fsk = layerP(l, 1);
    const float* Wr  = layerP(l, 2);
    const float* Fr  = layerP(l, 3);
    const float* Fb  = layerP(l, 4);

    // fused weights (transposed): rows = output col, ldt = din
    copy_cast_wT_k<<<g1(din * 64), TB, 0, stream>>>(Wsk, din, 64, Wcat, din, 0, 64);
    copy_cast_wT_k<<<g1(din * 128), TB, 0, stream>>>(Fsk, din, 128, Wcat, din, 64, 128);
    for (int r = 0; r < kR; ++r)
      copy_cast_wT_k<<<g1(din * 64), TB, 0, stream>>>(Wr + (size_t)r * din * 64, din, 64,
                                                      Wcat, din, 192 + r * 64, 64);
    for (int r = 0; r < kR; ++r)
      copy_cast_wT_k<<<g1(din * 128), TB, 0, stream>>>(Fr + (size_t)r * din * 128, din, 128,
                                                       Wcat, din, 384 + r * 128, 128);

    gemm(stream, xb, din, Wcat, din, Y, kYW, kN, kYW, din);

    film_post_k<<<g1(kN * kH), TB, 0, stream>>>(Y, nr, l * kH);

    fill_k<<<g1(kN * kR * 64), TB, 0, stream>>>(sums, 0.f, kN * kR * 64);
    fill_k<<<g1(kN * kR), TB, 0, stream>>>(cnt, 0.f, kN * kR);
    edge_msg_k<<<ceil_div(kE, 4), 256, 0, stream>>>(edge_index, edge_type, Y, Fb, sums, cnt);
    film_mean_k<<<g1(kN * kH), TB, 0, stream>>>(sums, cnt, nr, l * kH);

    if (l < kL) {   // graph-LN into next layer's bf16 input
      fill_k<<<1, 32, 0, stream>>>(red, 0.f, 2);
      reduce_sum_sq_k<<<256, 256, 0, stream>>>(nr, l * kH, red);
      ln_apply_k<<<g1(kN * kH), TB, 0, stream>>>(nr, l * kH, lnP(l, 0), lnP(l, 1), red, xb);
    }
  }

  // ================= pooling =================
  cast_f32_bf16_k<<<g1(kN * kNodeDim), TB, 0, stream>>>(nr, nrb, kN * kNodeDim);

  // layer 1: fused 832->512 weights (transposed 512 x 832) [s_sm | t_sm | s_sg | t_sg]
  const float* W0[4] = { aggP(0,0,0), aggP(0,1,0), aggP(1,0,0), aggP(1,1,0) };
  const float* B0[4] = { aggP(0,0,3), aggP(0,1,3), aggP(1,0,3), aggP(1,1,3) };
  for (int i = 0; i < 4; ++i)
    copy_cast_wT_k<<<g1(kNodeDim * 128), TB, 0, stream>>>(W0[i], kNodeDim, 128,
                                                          Wagg, kNodeDim, i * 128, 128);
  gemm(stream, nrb, kNodeDim, Wagg, kNodeDim, Y1, 512, kN, 512, kNodeDim);
  bias_act_cast4_k<<<g1(kN * 512), TB, 0, stream>>>(Y1, B0[0], B0[1], B0[2], B0[3], ag1b);

  // layer 2: four independent 128x128 blocks (each stored transposed 128 x 128)
  const float* W1[4] = { aggP(0,0,1), aggP(0,1,1), aggP(1,0,1), aggP(1,1,1) };
  const float* B1[4] = { aggP(0,0,4), aggP(0,1,4), aggP(1,0,4), aggP(1,1,4) };
  for (int i = 0; i < 4; ++i)
    copy_cast_wT_k<<<g1(128 * 128), TB, 0, stream>>>(W1[i], 128, 128,
                                                     Wagg + (size_t)i * 16384, 128, 0, 128);
  for (int i = 0; i < 4; ++i)
    gemm(stream, ag1b + i * 128, 512, Wagg + (size_t)i * 16384, 128, Y1 + i * 128, 512,
         kN, 128, 128);
  bias_act_cast4_k<<<g1(kN * 512), TB, 0, stream>>>(Y1, B1[0], B1[1], B1[2], B1[3], ag2b);

  // layer 3 scoring (128 -> 8, padded to 32): softmax at S[:,0:8], sigmoid at S[:,32:40]
  copy_cast_wT_k<<<g1(128 * 32), TB, 0, stream>>>(aggP(0,0,2), 128, 8, Wagg, 128, 0, 32);
  gemm(stream, ag2b + 0, 512, Wagg, 128, S + 0, 64, kN, 32, 128);
  copy_cast_wT_k<<<g1(128 * 32), TB, 0, stream>>>(aggP(1,0,2), 128, 8, Wagg, 128, 0, 32);
  gemm(stream, ag2b + 256, 512, Wagg, 128, S + 32, 64, kN, 32, 128);

  // segment softmax weights + sigmoid weights
  fill_k<<<g1(kB * kHeads), TB, 0, stream>>>(segmax, -3.402823466e38f, kB * kHeads);
  fill_k<<<g1(kB * kHeads), TB, 0, stream>>>(segsum, 0.f, kB * kHeads);
  seg_max_k<<<g1(kN * kHeads), TB, 0, stream>>>(S, aggP(0,0,5), batch, segmax);
  seg_expsum_k<<<g1(kN * kHeads), TB, 0, stream>>>(S, aggP(0,0,5), batch, segmax, wts, segsum);
  seg_norm_k<<<g1(kN * kHeads), TB, 0, stream>>>(wts, batch, segsum);
  sigmoid_w_k<<<g1(kN * kHeads), TB, 0, stream>>>(S, aggP(1,0,5), wts);

  fill_k<<<g1(kB * 2 * kGRS), TB, 0, stream>>>(out, 0.f, kB * 2 * kGRS);

  // transform (128 -> 416, transposed 416 x 128) + weighted scatter; T reuses nrb's region
  copy_cast_wT_k<<<g1(128 * kGRS), TB, 0, stream>>>(aggP(0,1,2), 128, kGRS, Wagg, 128, 0, kGRS);
  gemm(stream, ag2b + 128, 512, Wagg, 128, T, kGRS, kN, kGRS, 128);
  scatter_k<<<g1(kN * kGRS), TB, 0, stream>>>(T, aggP(0,1,5), wts, 0, batch, out, 0);

  copy_cast_wT_k<<<g1(128 * kGRS), TB, 0, stream>>>(aggP(1,1,2), 128, kGRS, Wagg, 128, 0, kGRS);
  gemm(stream, ag2b + 384, 512, Wagg, 128, T, kGRS, kN, kGRS, 128);
  scatter_k<<<g1(kN * kGRS), TB, 0, stream>>>(T, aggP(1,1,5), wts, 8, batch, out, kGRS);
}
